// TransformerBlock_6863357739126
// MI455X (gfx1250) — matrix operand order repaired and checked by reading, not compiled, not run on hardware
//
#include <hip/hip_runtime.h>
#include <cmath>
#include <cstdint>

typedef __attribute__((ext_vector_type(16))) __bf16 v16bf;
typedef __attribute__((ext_vector_type(8)))  float  v8f;

union BF16x16 { uint32_t u[8]; v16bf v; };

#define T_STEPS 10
#define N_TOK   256
#define DIM     4096
#define HIDDEN  14336
#define NH      32
#define NKV     8
#define HD      128
#define KV_DIM  (NKV * HD)   // 1024

// ---------------------------------------------------------------------------
// IFNode analytics, branchless. Constant drive y: spike at step t iff
// t % n == 0, n = ceil(1/y)  <=>  y in [1/n, 1/(n-1)).  Everything reduces to
// an interval ladder of compares -> v_cndmask chains (no division, no branch).
// ---------------------------------------------------------------------------
__device__ __forceinline__ float spike_mean(float y) {
  // #spikes in 10 steps = floor(10/n): 10,5,3,2,2,1..1,0
  float c = (y >= 0.1f)        ? 1.0f  : 0.0f;
  c       = (y >= 0.2f)        ? 2.0f  : c;
  c       = (y >= 0.33333334f) ? 3.0f  : c;
  c       = (y >= 0.5f)        ? 5.0f  : c;
  c       = (y >= 1.0f)        ? 10.0f : c;
  return c * 0.1f;
}
__device__ __forceinline__ uint32_t spike_mask(float y) {
  // bit (t-1) set iff spike at step t, t = 1..10
  uint32_t m = (y >= 0.1f)        ? 0x200u : 0u;    // n=10
  m          = (y >= 0.11111111f) ? 0x100u : m;     // n=9
  m          = (y >= 0.125f)      ? 0x080u : m;     // n=8
  m          = (y >= 0.14285715f) ? 0x040u : m;     // n=7
  m          = (y >= 0.16666667f) ? 0x020u : m;     // n=6
  m          = (y >= 0.2f)        ? 0x210u : m;     // n=5: t=5,10
  m          = (y >= 0.25f)       ? 0x088u : m;     // n=4: t=4,8
  m          = (y >= 0.33333334f) ? 0x124u : m;     // n=3: t=3,6,9
  m          = (y >= 0.5f)        ? 0x2AAu : m;     // n=2: t=2,4,..,10
  m          = (y >= 1.0f)        ? 0x3FFu : m;     // n=1: every step
  return m;
}

// relu(+x)/relu(-x) on a packed pair of bf16 in one dword: single VOP3P
// packed max-number each (CDNA5 V_PK_MAX_NUM_BF16), source-negate for -x.
__device__ __forceinline__ void relu_pair(uint32_t u, uint32_t& p, uint32_t& n) {
  asm("v_pk_max_num_bf16 %0, %1, 0" : "=v"(p) : "v"(u));
  asm("v_pk_max_num_bf16 %0, %1, 0 neg_lo:[1,0] neg_hi:[1,0]" : "=v"(n) : "v"(u));
}

// ---------------------------------------------------------------------------
// RMSNorm: one block per row of [256, 4096]
// ---------------------------------------------------------------------------
__global__ __launch_bounds__(256)
void rmsnorm_kernel(const float* __restrict__ x, const float* __restrict__ w,
                    float* __restrict__ out) {
  const int row = blockIdx.x;
  const int tid = threadIdx.x;
  const float* xr = x + (size_t)row * DIM;
  float ss = 0.0f;
#pragma unroll
  for (int i = 0; i < DIM / 256; ++i) {
    float v = xr[tid + i * 256];
    ss += v * v;
  }
#pragma unroll
  for (int o = 16; o > 0; o >>= 1) ss += __shfl_down(ss, o, 32);
  __shared__ float red[8];
  __shared__ float sscale;
  if ((tid & 31) == 0) red[tid >> 5] = ss;
  __syncthreads();
  if (tid == 0) {
    float t = 0.0f;
#pragma unroll
    for (int i = 0; i < 8; ++i) t += red[i];
    sscale = rsqrtf(t / (float)DIM + 1e-5f);
  }
  __syncthreads();
  const float s = sscale;
#pragma unroll
  for (int i = 0; i < DIM / 256; ++i) {
    int c = tid + i * 256;
    out[(size_t)row * DIM + c] = xr[c] * s * w[c];
  }
}

// ---------------------------------------------------------------------------
// bf16 WMMA GEMM: C = act(A) @ W^T.  A:[256,K] f32, W:[N,K] f32 row-major.
// BM=256 (full M: every weight byte fetched from HBM exactly once),
// BN=32, BK=32, 8 waves each owning a 32x32 output tile.
// Double-buffered LDS; next tile's global loads issue before current WMMAs.
// A staged once as *signed* bf16; relu+/relu- fragments derived with
// v_pk_max_num_bf16 at fragment-load time (DUAL shares W tiles and A LDS).
// MODE 0: out0 f32  = C + resid
// MODE 1: out0 u32  = spike_mask(Cp) | spike_mask(Cn) << 16   (DUAL)
// MODE 2: out0 f32  = smean(Cp) - smean(Cn)                   (DUAL)
// MODE 3: out0 f32  = smean(Cp) * aux
// ---------------------------------------------------------------------------
template<bool DUAL, bool RELU, int MODE>
__global__ __launch_bounds__(256)
void gemm_wmma(const float* __restrict__ A, const float* __restrict__ W,
               int N, int K,
               void* __restrict__ out0,
               const float* __restrict__ aux, const float* __restrict__ resid) {
  constexpr int BM = 256, BN = 32, BK = 32, LDW = BK + 8;
  __shared__ __bf16 As[2][BM][LDW];
  __shared__ __bf16 Bs[2][BN][LDW];

  const int tid   = threadIdx.x;
  const int lane  = tid & 31;
  const int wm    = tid >> 5;        // wave id = M strip (0..7), rows wm*32..+31
  const int n0    = blockIdx.x * BN;
  const int lr    = lane & 15;
  const int khalf = lane >> 4;

  const v8f zero = {0.f, 0.f, 0.f, 0.f, 0.f, 0.f, 0.f, 0.f};
  v8f accp[2][2] = {{zero, zero}, {zero, zero}};
  v8f accn[2][2] = {{zero, zero}, {zero, zero}};

  float4 areg[8];
  float4 breg;

  // prologue: fetch tile 0
#pragma unroll
  for (int it = 0; it < 8; ++it) {
    int s = tid + it * 256;                    // 2048 float4 slots (256x32)
    areg[it] = *(const float4*)(A + (size_t)(s >> 3) * K + ((s & 7) << 2));
  }
  breg = *(const float4*)(W + (size_t)(n0 + (tid >> 3)) * K + ((tid & 7) << 2));

  int buf = 0;
  for (int kt = 0; kt < K; kt += BK) {
    // ---- store fetched tile into LDS buffer `buf` (f32 -> signed bf16) ----
#pragma unroll
    for (int it = 0; it < 8; ++it) {
      int s = tid + it * 256;
      int row = s >> 3, c = (s & 7) << 2;
      As[buf][row][c + 0] = (__bf16)areg[it].x;
      As[buf][row][c + 1] = (__bf16)areg[it].y;
      As[buf][row][c + 2] = (__bf16)areg[it].z;
      As[buf][row][c + 3] = (__bf16)areg[it].w;
    }
    {
      int row = tid >> 3, c = (tid & 7) << 2;
      Bs[buf][row][c + 0] = (__bf16)breg.x;
      Bs[buf][row][c + 1] = (__bf16)breg.y;
      Bs[buf][row][c + 2] = (__bf16)breg.z;
      Bs[buf][row][c + 3] = (__bf16)breg.w;
    }
    __syncthreads();

    // ---- prefetch next tile while this tile's WMMAs run ----
    const bool has_next = (kt + BK) < K;
    if (has_next) {
#pragma unroll
      for (int it = 0; it < 8; ++it) {
        int s = tid + it * 256;
        areg[it] = *(const float4*)(A + (size_t)(s >> 3) * K + kt + BK + ((s & 7) << 2));
      }
      breg = *(const float4*)(W + (size_t)(n0 + (tid >> 3)) * K + kt + BK + ((tid & 7) << 2));
    }

    // ---- B fragments: lane holds column n = lr, K = khalf*16 + e ----
    BF16x16 bfr[2];
#pragma unroll
    for (int j = 0; j < 2; ++j) {
      const uint32_t* bp = (const uint32_t*)&Bs[buf][j * 16 + lr][0];
#pragma unroll
      for (int e = 0; e < 8; ++e) bfr[j].u[e] = bp[khalf * 8 + e];
    }
#pragma unroll
    for (int i = 0; i < 2; ++i) {
      // A fragment dwords: K = khalf*16 + e (contiguous, matches B)
      const uint32_t* ap = (const uint32_t*)&As[buf][wm * 32 + i * 16 + lr][0];
      uint32_t raw[8];
#pragma unroll
      for (int e = 0; e < 4; ++e) {
        raw[e]     = ap[khalf * 8 + e];
        raw[4 + e] = ap[khalf * 8 + 4 + e];
      }
      BF16x16 P, Nn;
      if (RELU) {
#pragma unroll
        for (int e = 0; e < 8; ++e) relu_pair(raw[e], P.u[e], Nn.u[e]);
      } else {
#pragma unroll
        for (int e = 0; e < 8; ++e) P.u[e] = raw[e];
      }
#pragma unroll
      for (int j = 0; j < 2; ++j)
        accp[i][j] = __builtin_amdgcn_wmma_f32_16x16x32_bf16(
            false, P.v, false, bfr[j].v, (short)0, accp[i][j], false, false);
      if constexpr (DUAL) {
#pragma unroll
        for (int j = 0; j < 2; ++j)
          accn[i][j] = __builtin_amdgcn_wmma_f32_16x16x32_bf16(
              false, Nn.v, false, bfr[j].v, (short)0, accn[i][j], false, false);
      }
    }
    __syncthreads();
    buf ^= 1;
  }

  // ---- epilogue: D layout -> VGPR r holds M = r + 8*khalf, N = lr ----
#pragma unroll
  for (int i = 0; i < 2; ++i) {
#pragma unroll
    for (int j = 0; j < 2; ++j) {
#pragma unroll
      for (int r = 0; r < 8; ++r) {
        int gm = wm * 32 + i * 16 + khalf * 8 + r;
        int gn = n0 + j * 16 + lr;
        size_t idx = (size_t)gm * N + gn;
        float vp = accp[i][j][r];
        if (MODE == 0) {
          ((float*)out0)[idx] = vp + (resid ? resid[idx] : 0.0f);
        } else if (MODE == 1) {
          ((uint32_t*)out0)[idx] =
              spike_mask(vp) | (spike_mask(accn[i][j][r]) << 16);
        } else if (MODE == 2) {
          ((float*)out0)[idx] = spike_mean(vp) - spike_mean(accn[i][j][r]);
        } else {
          ((float*)out0)[idx] = spike_mean(vp) * aux[idx];
        }
      }
    }
  }
}

// ---------------------------------------------------------------------------
// Per-token spiking head-mixing "attention", time-averaged with the source's
// transpose(-1,-2).reshape layout: xam[tok, d*32 + h] = mean_t xa[t,h,d].
// Spikes reconstructed per timestep from packed bitmasks (pos|neg<<16).
// ---------------------------------------------------------------------------
__global__ __launch_bounds__(256)
void attention_kernel(const uint32_t* __restrict__ mq,
                      const uint32_t* __restrict__ mk,
                      const uint32_t* __restrict__ mv,
                      float* __restrict__ xam) {
  const int tok = blockIdx.x;
  const int tid = threadIdx.x;
  __shared__ uint32_t Mq[DIM];
  __shared__ uint32_t Mk[KV_DIM], Mv[KV_DIM];
  __shared__ int8_t sq[DIM];
  __shared__ int8_t sk[KV_DIM], sv[KV_DIM];
  __shared__ float  attn[NH * NH];

  for (int i = tid; i < DIM; i += 256) Mq[i] = mq[(size_t)tok * DIM + i];
  for (int i = tid; i < KV_DIM; i += 256) {
    Mk[i] = mk[(size_t)tok * KV_DIM + i];
    Mv[i] = mv[(size_t)tok * KV_DIM + i];
  }
  float acc[16];
#pragma unroll
  for (int e = 0; e < 16; ++e) acc[e] = 0.0f;
  const int h  = tid >> 3;          // head 0..31
  const int d0 = (tid & 7) << 4;    // 16-wide d slice
  __syncthreads();

  for (int t = 0; t < T_STEPS; ++t) {     // bit index t = step-1
    for (int i = tid; i < DIM; i += 256) {
      uint32_t m = Mq[i];
      sq[i] = (int8_t)((int)((m >> t) & 1u) - (int)((m >> (t + 16)) & 1u));
    }
    for (int i = tid; i < KV_DIM; i += 256) {
      uint32_t a = Mk[i], b = Mv[i];
      sk[i] = (int8_t)((int)((a >> t) & 1u) - (int)((a >> (t + 16)) & 1u));
      sv[i] = (int8_t)((int)((b >> t) & 1u) - (int)((b >> (t + 16)) & 1u));
    }
    __syncthreads();
    // scores[i][j] = sum_d sq[i,d]*sk[j%8,d] * 0.125 : 4 scores/thread
#pragma unroll
    for (int s = 0; s < 4; ++s) {
      int sidx = (tid << 2) + s;
      int qi = sidx >> 5, kj = sidx & 31;
      const int8_t* q = &sq[qi * HD];
      const int8_t* k = &sk[(kj & 7) * HD];   // kv-head tile: j -> j % 8
      int dot = 0;
#pragma unroll 16
      for (int d = 0; d < HD; ++d) dot += (int)q[d] * (int)k[d];
      attn[sidx] = (float)dot * 0.125f;
    }
    __syncthreads();
    // xa[h, d0..d0+15] += attn[h,:] @ sv[(:%8), d0..]
#pragma unroll
    for (int j = 0; j < NH; ++j) {
      float a = attn[h * NH + j];
      const int8_t* v = &sv[(j & 7) * HD + d0];
#pragma unroll
      for (int e = 0; e < 16; ++e) acc[e] += a * (float)v[e];
    }
    __syncthreads();
  }
  // transposed reshape: c = d*32 + h ; time mean
#pragma unroll
  for (int e = 0; e < 16; ++e)
    xam[(size_t)tok * DIM + (size_t)(d0 + e) * NH + h] = acc[e] * (1.0f / T_STEPS);
}

// ---------------------------------------------------------------------------
// Host launcher
// ---------------------------------------------------------------------------
extern "C" void kernel_launch(void* const* d_in, const int* in_sizes, int n_in,
                              void* d_out, int out_size, void* d_ws, size_t ws_size,
                              hipStream_t stream) {
  (void)in_sizes; (void)n_in; (void)out_size; (void)ws_size;
  const float* x   = (const float*)d_in[0];
  // d_in[1] = freqs_cis: zeros, unused by the reference math
  const float* anw = (const float*)d_in[2];
  const float* fnw = (const float*)d_in[3];
  const float* wq  = (const float*)d_in[4];
  const float* wk  = (const float*)d_in[5];
  const float* wv  = (const float*)d_in[6];
  const float* wo  = (const float*)d_in[7];
  const float* w1  = (const float*)d_in[8];
  const float* w2  = (const float*)d_in[9];
  const float* w3  = (const float*)d_in[10];
  float* out = (float*)d_out;

  char* p = (char*)d_ws;
  auto take = [&](size_t bytes) {
    char* r = p;
    p += (bytes + 255) & ~(size_t)255;
    return r;
  };
  const size_t S1 = (size_t)N_TOK * DIM;      // 1,048,576
  const size_t S2 = (size_t)N_TOK * KV_DIM;   //   262,144
  const size_t S3 = (size_t)N_TOK * HIDDEN;   // 3,670,016

  float*    h1  = (float*)take(S1 * 4);
  float*    h   = (float*)take(S1 * 4);
  float*    h2  = (float*)take(S1 * 4);
  float*    xam = (float*)take(S1 * 4);
  float*    t3  = (float*)take(S3 * 4);
  float*    g   = (float*)take(S3 * 4);
  uint32_t* mq  = (uint32_t*)take(S1 * 4);
  uint32_t* mk  = (uint32_t*)take(S2 * 4);
  uint32_t* mv  = (uint32_t*)take(S2 * 4);

  const dim3 blk(256);
  const dim3 gDim(DIM / 32);
  const dim3 gKV(KV_DIM / 32);
  const dim3 gHid(HIDDEN / 32);

  // 1) h1 = rmsnorm(x, attn_norm_w)
  rmsnorm_kernel<<<N_TOK, blk, 0, stream>>>(x, anw, h1);
  // 2-4) q/k/v: dual relu GEMM -> packed spike bitmasks
  gemm_wmma<true, true, 1><<<gDim, blk, 0, stream>>>(h1, wq, DIM, DIM, mq, nullptr, nullptr);
  gemm_wmma<true, true, 1><<<gKV,  blk, 0, stream>>>(h1, wk, KV_DIM, DIM, mk, nullptr, nullptr);
  gemm_wmma<true, true, 1><<<gKV,  blk, 0, stream>>>(h1, wv, KV_DIM, DIM, mv, nullptr, nullptr);
  // 5) time-averaged spiking head-mixing attention (transposed layout)
  attention_kernel<<<N_TOK, blk, 0, stream>>>(mq, mk, mv, xam);
  // 6) h = xam @ wo^T + x
  gemm_wmma<false, false, 0><<<gDim, blk, 0, stream>>>(xam, wo, DIM, DIM, h, nullptr, x);
  // 7) h2 = rmsnorm(h, ffn_norm_w)
  rmsnorm_kernel<<<N_TOK, blk, 0, stream>>>(h, fnw, h2);
  // 8) t3 = smean(relu(h2)@w3^T) - smean(relu(-h2)@w3^T)
  gemm_wmma<true, true, 2><<<gHid, blk, 0, stream>>>(h2, w3, HIDDEN, DIM, t3, nullptr, nullptr);
  // 9) g = smean(relu(h2)@w1^T) * t3
  gemm_wmma<false, true, 3><<<gHid, blk, 0, stream>>>(h2, w1, HIDDEN, DIM, g, t3, nullptr);
  // 10) out = g @ w2^T + h
  gemm_wmma<false, false, 0><<<gDim, blk, 0, stream>>>(g, w2, DIM, HIDDEN, out, nullptr, h);
}
